// CognitiveAgent_55027120996869
// MI455X (gfx1250) — matrix operand order repaired and checked by reading, not compiled, not run on hardware
//
#include <hip/hip_runtime.h>
#include <hip/hip_bf16.h>
#include <math.h>

#define WS_B 32
#define WS_D 128
#define WS_K 16

typedef __attribute__((ext_vector_type(16))) __bf16 bf16x16;
typedef __attribute__((ext_vector_type(8)))  float  f32x8;
typedef __attribute__((ext_vector_type(4)))  float  f32x4;

union BfU {
  bf16x16 v;
  unsigned short us[16];
  f32x4 f4[2];
};

__device__ __forceinline__ unsigned short bf16bits(float f) {
  __bf16 h = (__bf16)f;  // native f32->bf16 (RNE) convert
  return __builtin_bit_cast(unsigned short, h);
}

// Register-resident top-K insert: cheap compare fast path, O(K) rare path.
__device__ __forceinline__ void topk_insert(float s, int id, float (&bv)[WS_K],
                                            int (&bi)[WS_K], float& kth) {
  if (s > kth) {
    int mi = 0;
    float mv = bv[0];
#pragma unroll
    for (int i = 1; i < WS_K; ++i)
      if (bv[i] < mv) { mv = bv[i]; mi = i; }
#pragma unroll
    for (int i = 0; i < WS_K; ++i)
      if (i == mi) { bv[i] = s; bi[i] = id; }
    float nm = bv[0];
#pragma unroll
    for (int i = 1; i < WS_K; ++i) nm = fminf(nm, bv[i]);
    kth = nm;
  }
}

// ---------------------------------------------------------------------------
// Kernel 1: q = query @ Wq^T + bq ; per-query norms ; bf16 A-fragments in the
// 16x32 per-lane layout with contiguous k (k = 16*(lane/16) + s), one fragment per K-chunk.
// ---------------------------------------------------------------------------
__global__ void __launch_bounds__(128)
qprep_kernel(const float* __restrict__ query, const float* __restrict__ Wq,
             const float* __restrict__ bq, float* __restrict__ qn2g,
             float* __restrict__ qng, unsigned short* __restrict__ afrag) {
  __shared__ float lq[WS_B * WS_D];   // input queries
  __shared__ float lqq[WS_B * WS_D];  // projected queries
  const int tid = threadIdx.x;  // 128 threads

  for (int i = tid; i < WS_B * WS_D; i += 128) lq[i] = query[i];
  __syncthreads();

  // thread d computes column d of q for all 32 queries
  const int d = tid;
  const float bqa = bq[d];
  for (int b = 0; b < WS_B; ++b) {
    float acc = bqa;
    for (int k = 0; k < WS_D; ++k) acc += lq[b * WS_D + k] * Wq[d * WS_D + k];
    lqq[b * WS_D + d] = acc;
  }
  __syncthreads();

  if (tid < WS_B) {
    float s = 0.0f;
    for (int dd = 0; dd < WS_D; ++dd) {
      float v = lqq[tid * WS_D + dd];
      s += v * v;
    }
    qn2g[tid] = s;
    qng[tid] = sqrtf(s);
  }

  // A-fragments: entries (mt in 0..1, c in 0..3, lane in 0..31), 16 bf16 each.
  for (int e = tid; e < 256; e += 128) {
    const int mt = e >> 7;
    const int c = (e >> 5) & 3;
    const int lane = e & 31;
    const int m = mt * 16 + (lane & 15);
    const int kb = c * 32;
    const int koff = (lane < 16) ? 0 : 16;
    unsigned short* dst = afrag + ((((mt * 4 + c) * 32) + lane) << 4);
#pragma unroll
    for (int el = 0; el < 16; ++el) {
      const int k = kb + koff + el;
      dst[el] = bf16bits(lqq[m * WS_D + k]);
    }
  }
}

// ---------------------------------------------------------------------------
// Kernel 2: stream memory, WMMA dot tiles, fused score + per-lane top-K +
// min-d2; per-block merge -> workspace.
// ---------------------------------------------------------------------------
__global__ void __launch_bounds__(256)
score_kernel(const float* __restrict__ memory, const float* __restrict__ importance,
             const int* __restrict__ timestamps, const int* __restrict__ ctp,
             const float* __restrict__ qn2g, const float* __restrict__ qng,
             const unsigned short* __restrict__ afrag,
             float* __restrict__ blkScore, int* __restrict__ blkIdx,
             float* __restrict__ blkMin, int N, int G) {
  __shared__ float ldsDot[8][16][20];   // [wave][query][row], padded stride
  __shared__ f32x4 ldsMeta[8][16];      // {mn, mn2, c0, c1} per row
  __shared__ float mS[8][32][WS_K];
  __shared__ int   mI[8][32][WS_K];
  __shared__ float mM[8][32];

  const int tid = threadIdx.x;
  const int wave = tid >> 5;
  const int lane = tid & 31;
  const int mt = wave >> 2;     // query tile (0: q0-15, 1: q16-31)
  const int j = wave & 3;       // row-tile sub-stream within block
  const int khalf = lane >> 4;  // which K-half of the B fragment this lane holds
  const int nloc = lane & 15;   // row-in-tile (loader) / query-in-tile (selector)
  const int g = blockIdx.x;
  const float invCT = 1.0f / ((float)ctp[0] + 1.0f);

  // Load pre-swizzled A fragments (bf16, WMMA layout): 4 K-chunks.
  BfU A[4];
#pragma unroll
  for (int c = 0; c < 4; ++c) {
    const f32x4* p = (const f32x4*)(afrag + ((((mt * 4 + c) * 32) + lane) << 4));
    A[c].f4[0] = p[0];
    A[c].f4[1] = p[1];
  }

  const float qn = qng[mt * 16 + nloc];
  const float qn2 = qn2g[mt * 16 + nloc];

  float best[WS_K];
  int bidx[WS_K];
#pragma unroll
  for (int i = 0; i < WS_K; ++i) { best[i] = -__builtin_inff(); bidx[i] = 0; }
  float kth = -__builtin_inff();
  float mind2 = __builtin_inff();

  const int T = (N + 15) >> 4;  // 16-row tiles
  const int stride = G << 2;    // 4 wave-pairs per block

  for (int t = g * 4 + j; t < T; t += stride) {
    const int base = t << 4;
    const int row = base + nloc;
    const int rowc = (row < N) ? row : (N - 1);
    const float* rowp = memory + (size_t)rowc * WS_D + khalf * 16;

    {  // prefetch whole next grid-stride tile (2 x 128B per lane = 8KB)
      const int tn = t + stride;
      if (tn < T) {
        const float* pf = memory + ((size_t)(tn << 4)) * WS_D;
        __builtin_prefetch(pf + lane * 32, 0, 0);
        __builtin_prefetch(pf + 1024 + lane * 32, 0, 0);
      }
    }

    float ssq = 0.0f;
    f32x8 acc = {};
#pragma unroll
    for (int c = 0; c < 4; ++c) {
      BfU Bf;
#pragma unroll
      for (int i = 0; i < 4; ++i) {
        f32x4 x = *(const f32x4*)(rowp + c * 32 + 4 * i);
        ssq += x[0] * x[0] + x[1] * x[1] + x[2] * x[2] + x[3] * x[3];
        Bf.v[i * 4 + 0] = (__bf16)x[0];
        Bf.v[i * 4 + 1] = (__bf16)x[1];
        Bf.v[i * 4 + 2] = (__bf16)x[2];
        Bf.v[i * 4 + 3] = (__bf16)x[3];
      }
      acc = __builtin_amdgcn_wmma_f32_16x16x32_bf16(
          false, A[c].v, false, Bf.v, (short)0, acc, false, false);
    }

    // Per-row metadata (lanes 0-15 own the 16 rows of this tile).
    const float full = ssq + __shfl_xor(ssq, 16, 32);
    if (lane < 16) {
      const float imp = importance[rowc];
      const float ts = (float)timestamps[rowc];
      const float wimp = 0.5f + 0.5f * imp;
      f32x4 mv;
      mv[0] = __builtin_amdgcn_sqrtf(full);
      mv[1] = full;
      mv[2] = 0.7f * wimp;
      mv[3] = 0.3f * wimp * (ts + 1.0f) * invCT;
      ldsMeta[wave][nloc] = mv;
    }

    // Spill dot tile: C layout -> VGPR jv: lanes0-15 M=jv, lanes16-31 M=jv+8.
#pragma unroll
    for (int jv = 0; jv < 8; ++jv)
      ldsDot[wave][jv + 8 * khalf][nloc] = acc[jv];

    __builtin_amdgcn_wave_barrier();
    asm volatile("s_wait_dscnt 0" ::: "memory");

    // Selection: lane owns query nloc; its K-half scans 8 of the 16 rows.
    const f32x4 d0 = *(const f32x4*)&ldsDot[wave][nloc][khalf * 8];
    const f32x4 d1 = *(const f32x4*)&ldsDot[wave][nloc][khalf * 8 + 4];
#pragma unroll
    for (int nn = 0; nn < 8; ++nn) {
      const int n = khalf * 8 + nn;
      const int idxg = base + n;
      const float dt = (nn < 4) ? d0[nn] : d1[nn - 4];
      const f32x4 md = ldsMeta[wave][n];
      const float denom = fmaxf(qn * md[0], 1e-8f);
      const float r = __builtin_amdgcn_rcpf(denom);
      const float sc = dt * md[2] * r + md[3];
      const float d2 = (qn2 + md[1]) - 2.0f * dt;
      if (idxg < N) {
        mind2 = fminf(mind2, d2);
        topk_insert(sc, idxg, best, bidx, kth);
      }
    }
    __builtin_amdgcn_wave_barrier();
  }

  // Block-level merge.
#pragma unroll
  for (int i = 0; i < WS_K; ++i) {
    mS[wave][lane][i] = best[i];
    mI[wave][lane][i] = bidx[i];
  }
  mM[wave][lane] = mind2;
  __syncthreads();

  if (tid < 32) {
    const int qg = tid;
    const int qmt = qg >> 4;
    const int qm = qg & 15;
    float mm = __builtin_inff();
#pragma unroll
    for (int ww = 0; ww < 4; ++ww) {
      mm = fminf(mm, mM[qmt * 4 + ww][qm]);
      mm = fminf(mm, mM[qmt * 4 + ww][qm + 16]);
    }
    blkMin[g * 32 + qg] = mm;

    for (int kk = 0; kk < WS_K; ++kk) {
      float bv = -__builtin_inff();
      int bw = 0, bl = 0, bc = 0;
      for (int ww = 0; ww < 4; ++ww)
        for (int hh = 0; hh < 2; ++hh)
          for (int cc = 0; cc < WS_K; ++cc) {
            const float s = mS[qmt * 4 + ww][qm + hh * 16][cc];
            if (s > bv) { bv = s; bw = qmt * 4 + ww; bl = qm + hh * 16; bc = cc; }
          }
      blkScore[(g * 32 + qg) * WS_K + kk] = bv;
      blkIdx[(g * 32 + qg) * WS_K + kk] = mI[bw][bl][bc];
      mS[bw][bl][bc] = -__builtin_inff();
    }
  }
}

// ---------------------------------------------------------------------------
// Kernel 3: final merge of G*16 candidates per query -> top-16 (descending),
// min-d2 -> novelty.
// ---------------------------------------------------------------------------
__global__ void __launch_bounds__(256)
reduce_kernel(const float* __restrict__ bs, const int* __restrict__ bi,
              const float* __restrict__ bm, int G,
              float* __restrict__ outS, int* __restrict__ outI,
              float* __restrict__ outN) {
  __shared__ float ls[256 * WS_K];
  __shared__ int li[256 * WS_K];
  __shared__ float red[256];
  __shared__ float l2s[256];
  __shared__ int l2i[256];

  const int b = blockIdx.x;
  const int tid = threadIdx.x;

  float best[WS_K];
  int bidv[WS_K];
#pragma unroll
  for (int i = 0; i < WS_K; ++i) { best[i] = -__builtin_inff(); bidv[i] = 0; }
  float kth = -__builtin_inff();

  const int total = G * WS_K;
  for (int c = tid; c < total; c += 256) {
    const int g = c >> 4;
    const int kk = c & 15;
    topk_insert(bs[(g * 32 + b) * WS_K + kk], bi[(g * 32 + b) * WS_K + kk],
                best, bidv, kth);
  }
  float m = __builtin_inff();
  for (int g = tid; g < G; g += 256) m = fminf(m, bm[g * 32 + b]);

#pragma unroll
  for (int i = 0; i < WS_K; ++i) {
    ls[tid * WS_K + i] = best[i];
    li[tid * WS_K + i] = bidv[i];
  }
  red[tid] = m;
  __syncthreads();

  for (int st = 128; st > 0; st >>= 1) {
    if (tid < st) red[tid] = fminf(red[tid], red[tid + st]);
    __syncthreads();
  }

  if (tid < 16) {
    float b2[WS_K];
    int i2[WS_K];
#pragma unroll
    for (int i = 0; i < WS_K; ++i) { b2[i] = -__builtin_inff(); i2[i] = 0; }
    float kth2 = -__builtin_inff();
    for (int src = 0; src < 16; ++src) {
      const int lst = tid * 16 + src;
      for (int cc = 0; cc < WS_K; ++cc)
        topk_insert(ls[lst * WS_K + cc], li[lst * WS_K + cc], b2, i2, kth2);
    }
#pragma unroll
    for (int i = 0; i < WS_K; ++i) {
      l2s[tid * WS_K + i] = b2[i];
      l2i[tid * WS_K + i] = i2[i];
    }
  }
  __syncthreads();

  if (tid == 0) {
    for (int kk = 0; kk < WS_K; ++kk) {
      float bv = -__builtin_inff();
      int slot = 0;
      for (int c = 0; c < 256; ++c)
        if (l2s[c] > bv) { bv = l2s[c]; slot = c; }
      outS[b * WS_K + kk] = bv;
      outI[b * WS_K + kk] = l2i[slot];
      l2s[slot] = -__builtin_inff();
    }
    const float d2 = fmaxf(red[0], 0.0f);
    outN[b] = fminf(1.0f, sqrtf(d2) * 0.1f);
  }
}

extern "C" void kernel_launch(void* const* d_in, const int* in_sizes, int n_in,
                              void* d_out, int out_size, void* d_ws, size_t ws_size,
                              hipStream_t stream) {
  (void)n_in; (void)out_size;
  const float* query = (const float*)d_in[0];
  const float* memory = (const float*)d_in[1];
  const float* importance = (const float*)d_in[2];
  const int* timestamps = (const int*)d_in[3];
  const float* Wq = (const float*)d_in[4];
  const float* bq = (const float*)d_in[5];
  const int* ctp = (const int*)d_in[6];
  const int N = in_sizes[1] / WS_D;

  char* ws = (char*)d_ws;
  float* qn2 = (float*)(ws + 0);                         // 128 B
  float* qng = (float*)(ws + 128);                       // 128 B
  unsigned short* afrag = (unsigned short*)(ws + 256);   // 8192 B
  const size_t fixed = 8448;
  const size_t per = (size_t)32 * WS_K * 4 * 2 + 32 * 4; // 4224 B / pass-1 block
  int G = 512;
  if (ws_size < fixed + (size_t)G * per) {
    G = (int)((ws_size - fixed) / per);
    if (G < 1) G = 1;
  }
  float* bs = (float*)(ws + fixed);
  int* bib = (int*)(ws + fixed + (size_t)G * 32 * WS_K * 4);
  float* bm = (float*)(ws + fixed + (size_t)G * 32 * WS_K * 8);

  qprep_kernel<<<1, 128, 0, stream>>>(query, Wq, bq, qn2, qng, afrag);
  score_kernel<<<G, 256, 0, stream>>>(memory, importance, timestamps, ctp,
                                      qn2, qng, afrag, bs, bib, bm, N, G);

  float* outS = (float*)d_out;
  int* outI = (int*)((float*)d_out + WS_B * WS_K);
  float* outN = (float*)d_out + 2 * WS_B * WS_K;
  reduce_kernel<<<WS_B, 256, 0, stream>>>(bs, bib, bm, G, outS, outI, outN);
}
